// QuantLinear_76888504533247
// MI455X (gfx1250) — 8-bit matrix operand order repaired and checked by reading, compiled, not run on hardware
//
#include <hip/hip_runtime.h>

typedef __attribute__((ext_vector_type(8))) int   v8i;
typedef __attribute__((ext_vector_type(4))) int   v4i;
typedef __attribute__((ext_vector_type(2))) int   v2i;
typedef __attribute__((ext_vector_type(8))) float v8f;
typedef __attribute__((ext_vector_type(4))) float v4f;

template <int QMAX>
__global__ __launch_bounds__(256) void quant_groups(
    const float* __restrict__ src, signed char* __restrict__ q,
    float* __restrict__ scale, int rows, int K, int G)
{
    const int gwave = (blockIdx.x * 256 + threadIdx.x) >> 5;
    const int lane  = threadIdx.x & 31;
    if (gwave >= rows * G) return;
    const int row = gwave / G;
    const int g   = gwave - row * G;

    const float* p = src + (size_t)row * K + (g << 7) + (lane << 2);
    v4f v = *(const v4f*)p;
    float amax = fmaxf(fmaxf(fabsf(v.x), fabsf(v.y)),
                       fmaxf(fabsf(v.z), fabsf(v.w)));
#pragma unroll
    for (int off = 16; off >= 1; off >>= 1)
        amax = fmaxf(amax, __shfl_xor(amax, off, 32));

    float s = amax * (1.0f / (float)QMAX);
    if (s == 0.0f) s = 1.0f;
    const float inv = 1.0f / s;

    int c0 = min(QMAX, max(-QMAX, (int)rintf(v.x * inv)));
    int c1 = min(QMAX, max(-QMAX, (int)rintf(v.y * inv)));
    int c2 = min(QMAX, max(-QMAX, (int)rintf(v.z * inv)));
    int c3 = min(QMAX, max(-QMAX, (int)rintf(v.w * inv)));
    int packed = (c0 & 255) | ((c1 & 255) << 8) | ((c2 & 255) << 16) | (c3 << 24);

    *(int*)(q + (size_t)row * K + (g << 7) + (lane << 2)) = packed;
    if (lane == 0) scale[(size_t)g * rows + row] = s;
}

__global__ __launch_bounds__(256) void gemm_q4q8(
    const signed char* __restrict__ qa, const signed char* __restrict__ qw,
    const float* __restrict__ saT, const float* __restrict__ swT,
    const float* __restrict__ bias, float* __restrict__ out,
    int M, int N, int K)
{
    const int G     = K >> 7;
    const int lane  = threadIdx.x & 31;
    const int wave  = threadIdx.x >> 5;
    const int l16   = lane & 15;
    const int lhalf = lane >> 4;

    const int m0 = blockIdx.x * 128 + (wave & 3) * 32;
    const int n0 = blockIdx.y * 128 + (wave >> 2) * 64;

    const signed char* aRow[2] = { qa + (size_t)(m0 + l16) * K,
                                   qa + (size_t)(m0 + 16 + l16) * K };
    const signed char* bRow[4];
#pragma unroll
    for (int nt = 0; nt < 4; ++nt)
        bRow[nt] = qw + (size_t)(n0 + (nt << 4) + l16) * K;

    v8f acc[2][4];
#pragma unroll
    for (int mt = 0; mt < 2; ++mt)
#pragma unroll
        for (int nt = 0; nt < 4; ++nt) acc[mt][nt] = (v8f)0.0f;

    for (int g = 0; g < G; ++g) {
        const int k0 = g << 7;

        __builtin_prefetch(aRow[0] + k0 + 128, 0, 1);
        __builtin_prefetch(bRow[0] + k0 + 128, 0, 1);
        __builtin_prefetch(bRow[2] + k0 + 128, 0, 1);

        v8i A[2][2];
#pragma unroll
        for (int mt = 0; mt < 2; ++mt)
#pragma unroll
            for (int kh = 0; kh < 2; ++kh) {
                const signed char* p = aRow[mt] + k0 + (kh << 6) + (lhalf << 3);
                v2i d0 = *(const v2i*)(p);
                v2i d1 = *(const v2i*)(p + 16);
                v2i d2 = *(const v2i*)(p + 32);
                v2i d3 = *(const v2i*)(p + 48);
                v8i a;
                a[0] = d0.x; a[1] = d0.y; a[2] = d1.x; a[3] = d1.y;
                a[4] = d2.x; a[5] = d2.y; a[6] = d3.x; a[7] = d3.y;
                A[mt][kh] = a;
            }

        v8i B[4][2];
#pragma unroll
        for (int nt = 0; nt < 4; ++nt)
#pragma unroll
            for (int kh = 0; kh < 2; ++kh) {
                const signed char* p = bRow[nt] + k0 + (kh << 6) + (lhalf << 3);
                v2i d0 = *(const v2i*)(p);
                v2i d1 = *(const v2i*)(p + 16);
                v2i d2 = *(const v2i*)(p + 32);
                v2i d3 = *(const v2i*)(p + 48);
                v8i b;
                b[0] = d0.x; b[1] = d0.y; b[2] = d1.x; b[3] = d1.y;
                b[4] = d2.x; b[5] = d2.y; b[6] = d3.x; b[7] = d3.y;
                B[nt][kh] = b;
            }

        v8f sA[2];
#pragma unroll
        for (int mt = 0; mt < 2; ++mt) {
            const float* sp = saT + (size_t)g * M + m0 + (mt << 4) + (lhalf << 3);
            v4f s0 = *(const v4f*)(sp);
            v4f s1 = *(const v4f*)(sp + 4);
            v8f s;
            s[0] = s0.x; s[1] = s0.y; s[2] = s0.z; s[3] = s0.w;
            s[4] = s1.x; s[5] = s1.y; s[6] = s1.z; s[7] = s1.w;
            sA[mt] = s;
        }
        float sB[4];
#pragma unroll
        for (int nt = 0; nt < 4; ++nt)
            sB[nt] = swT[(size_t)g * N + n0 + (nt << 4) + l16];

#pragma unroll
        for (int mt = 0; mt < 2; ++mt)
#pragma unroll
            for (int nt = 0; nt < 4; ++nt) {
                v8i c = (v8i)0;
                c = __builtin_amdgcn_wmma_i32_16x16x64_iu8(
                        true, A[mt][0], true, B[nt][0], c, false, false);
                c = __builtin_amdgcn_wmma_i32_16x16x64_iu8(
                        true, A[mt][1], true, B[nt][1], c, false, false);
#pragma unroll
                for (int r = 0; r < 8; ++r)
                    acc[mt][nt][r] = fmaf((float)c[r], sA[mt][r] * sB[nt],
                                          acc[mt][nt][r]);
            }
    }

    float bv[4];
#pragma unroll
    for (int nt = 0; nt < 4; ++nt) bv[nt] = bias[n0 + (nt << 4) + l16];

#pragma unroll
    for (int mt = 0; mt < 2; ++mt)
#pragma unroll
        for (int nt = 0; nt < 4; ++nt)
#pragma unroll
            for (int r = 0; r < 8; ++r) {
                const int m = m0 + (mt << 4) + (lhalf << 3) + r;
                const int n = n0 + (nt << 4) + l16;
                out[(size_t)m * N + n] = acc[mt][nt][r] + bv[nt];
            }
}

extern "C" void kernel_launch(void* const* d_in, const int* in_sizes, int n_in,
                              void* d_out, int out_size, void* d_ws, size_t ws_size,
                              hipStream_t stream)
{
    const float* x    = (const float*)d_in[0];
    const float* w    = (const float*)d_in[1];
    const float* bias = (const float*)d_in[2];
    float*       out  = (float*)d_out;

    const int N = in_sizes[2];
    const int K = in_sizes[1] / N;
    const int M = in_sizes[0] / K;
    const int G = K >> 7;

    char*        ws = (char*)d_ws;
    signed char* qa = (signed char*)ws;
    signed char* qw = qa + (size_t)M * K;
    float*       sa = (float*)(qw + (size_t)N * K);
    float*       sw = sa + (size_t)G * M;

    {
        const int waves  = M * G;
        const int blocks = (waves * 32 + 255) / 256;
        quant_groups<127><<<blocks, 256, 0, stream>>>(x, qa, sa, M, K, G);
    }
    {
        const int waves  = N * G;
        const int blocks = (waves * 32 + 255) / 256;
        quant_groups<7><<<blocks, 256, 0, stream>>>(w, qw, sw, N, K, G);
    }

    dim3 grid(M / 128, N / 128);
    gemm_q4q8<<<grid, 256, 0, stream>>>(qa, qw, sa, sw, bias, out, M, N, K);
}
